// GCN_24713241821268
// MI455X (gfx1250) — matrix operand order repaired and checked by reading, compiled, not run on hardware
//
#include <hip/hip_runtime.h>

// ---------------------------------------------------------------------------
// GCN layer: agg = scatter(norm * (x@W)) + b ; out = BN(relu(agg)) + x@W_res + b_res
// MI455X (gfx1250): wave32, WMMA bf16 16x16x32 with f32 accumulation.
// Edge phase is L2-resident (h + agg = 102MB < 192MB L2); elementwise kernels
// are float4-vectorized to be bandwidth- not issue-limited.
// ---------------------------------------------------------------------------

typedef __attribute__((ext_vector_type(16))) __bf16 v16bf;
typedef __attribute__((ext_vector_type(8)))  __bf16 v8bf;
typedef __attribute__((ext_vector_type(8)))  float  v8f;

#define IN_F   128
#define OUT_F  128
#define LDS_W  136   // 128 + 8 bf16 pad -> row stride 272B, conflict-free b128 LDS loads

union Frag { v16bf v; v8bf h[2]; };

// ---- Dual dense path: blockIdx.y==0 -> h = x@W ; blockIdx.y==1 -> res = x@W_res + b_res
__global__ __launch_bounds__(256)
void gcn_gemm_bf16(const float* __restrict__ x,
                   const float* __restrict__ W0,
                   const float* __restrict__ W1,
                   const float* __restrict__ bres,
                   float* __restrict__ h_out,
                   float* __restrict__ res_out,
                   int n) {
  __shared__ __bf16 lds[OUT_F * LDS_W + 64 * LDS_W];
  __bf16* wt = lds;                  // weights transposed: wt[col][k]
  __bf16* xs = lds + OUT_F * LDS_W;  // x tile: xs[r][k]

  const int tid   = threadIdx.x;
  const int which = blockIdx.y;
  const float* W  = which ? W1 : W0;
  float* out      = which ? res_out : h_out;
  const int m0    = blockIdx.x * 64;

  // Stage weights transposed (coalesced global read, bf16 convert).
  for (int idx = tid; idx < IN_F * OUT_F; idx += 256) {
    const int k = idx >> 7, col = idx & 127;
    wt[col * LDS_W + k] = (__bf16)W[idx];
  }
  // Stage x tile, zero-pad rows past n.
  for (int idx = tid; idx < 64 * IN_F; idx += 256) {
    const int r = idx >> 7, k = idx & 127;
    const int row = m0 + r;
    const float v = (row < n) ? x[(size_t)row * IN_F + k] : 0.0f;
    xs[r * LDS_W + k] = (__bf16)v;
  }
  __syncthreads();

  const int wave = tid >> 5;
  const int lane = tid & 31;
  const int half = lane >> 4;       // 0: lanes 0-15, 1: lanes 16-31
  const int l16  = lane & 15;
  const int rowTile = (wave & 3) * 16;   // 0/16/32/48 within the 64-row tile
  const int colHalf = (wave >> 2) * 64;  // 0 or 64

  v8f acc[4] = {};

  // K = 128 in 4 steps of 32. Uniform control flow -> EXEC all-ones for WMMA.
  for (int kc = 0; kc < 4; ++kc) {
    // A fragment (16x32 bf16): lane holds row M=l16;
    // half==0 -> K = kc*32 + {0..7, 16..23}; half==1 -> + {8..15, 24..31}.
    Frag a;
    const int m  = rowTile + l16;
    const int ka = kc * 32 + half * 8;
    a.h[0] = *(const v8bf*)&xs[m * LDS_W + ka];
    a.h[1] = *(const v8bf*)&xs[m * LDS_W + ka + 16];

    // B fragment (32x16 bf16): lane holds col N=l16;
    // half==0 -> K = kc*32 + {0..7, 16..23}; half==1 -> + {8..15, 24..31}.
    const int kb = kc * 32 + half * 8;
#pragma unroll
    for (int nt = 0; nt < 4; ++nt) {
      Frag b;
      const int col = colHalf + nt * 16 + l16;
      b.h[0] = *(const v8bf*)&wt[col * LDS_W + kb];
      b.h[1] = *(const v8bf*)&wt[col * LDS_W + kb + 16];
      acc[nt] = __builtin_amdgcn_wmma_f32_16x16x32_bf16(
          false, a.v, false, b.v, (short)0, acc[nt], false, false);
    }
  }

  // C/D layout: VGPR r -> lanes 0-15 hold M=r, lanes 16-31 hold M=8+r; N=l16.
#pragma unroll
  for (int nt = 0; nt < 4; ++nt) {
    const int col = colHalf + nt * 16 + l16;
    const float bv = which ? bres[col] : 0.0f;
#pragma unroll
    for (int r = 0; r < 8; ++r) {
      const int row = m0 + rowTile + half * 8 + r;
      if (row < n) out[(size_t)row * OUT_F + col] = acc[nt][r] + bv;
    }
  }
}

// ---- deg[i] = 1 (self loop) --------------------------------------------------
__global__ void deg_init(float* __restrict__ deg, int n) {
  const int i = blockIdx.x * blockDim.x + threadIdx.x;
  if (i < n) deg[i] = 1.0f;
}

// ---- stats zero --------------------------------------------------------------
__global__ void zero_stats(float* __restrict__ stats) {
  stats[threadIdx.x] = 0.0f;   // launched <<<1,256>>>
}

// ---- deg[col] += 1 per edge --------------------------------------------------
__global__ void deg_count(const long long* __restrict__ col,
                          float* __restrict__ deg, int nE) {
  const int e = blockIdx.x * blockDim.x + threadIdx.x;
  if (e < nE) atomicAdd(&deg[col[e]], 1.0f);
}

// ---- agg = b + self-loop message h[i]/deg[i]  (float4 per lane) --------------
__global__ void agg_init(const float* __restrict__ h,
                         const float* __restrict__ deg,
                         const float* __restrict__ b,
                         float* __restrict__ agg, int n) {
  const int i = blockIdx.x * blockDim.x + threadIdx.x;  // n*32 threads
  if (i >= n * 32) return;
  const int node = i >> 5;
  const int f4   = (i & 31) * 4;
  const float inv = 1.0f / deg[node];
  const float4 hv = *(const float4*)&h[(size_t)node * OUT_F + f4];
  const float4 bv = *(const float4*)&b[f4];
  float4 o;
  o.x = bv.x + hv.x * inv;
  o.y = bv.y + hv.y * inv;
  o.z = bv.z + hv.z * inv;
  o.w = bv.w + hv.w * inv;
  *(float4*)&agg[(size_t)node * OUT_F + f4] = o;
}

// ---- per-edge scatter: agg[col] += h[row] * dinv[row]*dinv[col] --------------
// One wave32 per edge, float4 per lane: b128 gather, 4 atomics per lane.
__global__ __launch_bounds__(256)
void edge_scatter(const long long* __restrict__ ei,
                  const float* __restrict__ h,
                  const float* __restrict__ deg,
                  float* __restrict__ agg, int nE) {
  const long long t = (long long)blockIdx.x * blockDim.x + threadIdx.x;
  const long long e = t >> 5;
  const int f4 = (int)(t & 31) * 4;
  if (e >= nE) return;
  const long long row = ei[e];        // source (wave-uniform)
  const long long col = ei[nE + e];   // destination (wave-uniform)
  const float norm = rsqrtf(deg[row]) * rsqrtf(deg[col]);
  const float4 hv = *(const float4*)&h[row * OUT_F + f4];
  float* dst = &agg[col * OUT_F + f4];
  atomicAdd(dst + 0, hv.x * norm);
  atomicAdd(dst + 1, hv.y * norm);
  atomicAdd(dst + 2, hv.z * norm);
  atomicAdd(dst + 3, hv.w * norm);
}

// ---- per-feature sum / sumsq of relu(agg), float4 per lane -------------------
__global__ __launch_bounds__(256)
void bn_stats(const float* __restrict__ agg, float* __restrict__ stats, int n) {
  const int g  = threadIdx.x >> 5;        // 8 row groups per block
  const int f4 = (threadIdx.x & 31) * 4;  // 32 lanes cover 128 features
  const int r0 = blockIdx.x * 256;
  int rend = r0 + 256; if (rend > n) rend = n;
  float4 s  = {0.f, 0.f, 0.f, 0.f};
  float4 s2 = {0.f, 0.f, 0.f, 0.f};
  for (int r = r0 + g; r < rend; r += 8) {
    float4 v = *(const float4*)&agg[(size_t)r * OUT_F + f4];
    v.x = fmaxf(v.x, 0.f); v.y = fmaxf(v.y, 0.f);
    v.z = fmaxf(v.z, 0.f); v.w = fmaxf(v.w, 0.f);
    s.x += v.x; s.y += v.y; s.z += v.z; s.w += v.w;
    s2.x += v.x * v.x; s2.y += v.y * v.y;
    s2.z += v.z * v.z; s2.w += v.w * v.w;
  }
  atomicAdd(&stats[f4 + 0], s.x);
  atomicAdd(&stats[f4 + 1], s.y);
  atomicAdd(&stats[f4 + 2], s.z);
  atomicAdd(&stats[f4 + 3], s.w);
  atomicAdd(&stats[128 + f4 + 0], s2.x);
  atomicAdd(&stats[128 + f4 + 1], s2.y);
  atomicAdd(&stats[128 + f4 + 2], s2.z);
  atomicAdd(&stats[128 + f4 + 3], s2.w);
}

// ---- out = gamma*(relu(agg)-mean)*rsqrt(var+eps) + beta + res (float4) -------
__global__ void finalize(const float* __restrict__ agg,
                         const float* __restrict__ res,
                         const float* __restrict__ stats,
                         const float* __restrict__ gamma,
                         const float* __restrict__ beta,
                         float* __restrict__ out, int n) {
  const int i = blockIdx.x * blockDim.x + threadIdx.x;  // n*32 threads
  if (i >= n * 32) return;
  const int node = i >> 5;
  const int f4   = (i & 31) * 4;
  const float invn = 1.0f / (float)n;
  const float4 sm  = *(const float4*)&stats[f4];
  const float4 sq  = *(const float4*)&stats[128 + f4];
  const float4 gm  = *(const float4*)&gamma[f4];
  const float4 bt  = *(const float4*)&beta[f4];
  const size_t off = (size_t)node * OUT_F + f4;
  float4 v = *(const float4*)&agg[off];
  const float4 rv = *(const float4*)&res[off];
  float4 o;
  {
    const float mean = sm.x * invn, var = sq.x * invn - mean * mean;
    const float r = fmaxf(v.x, 0.f);
    o.x = gm.x * (r - mean) * rsqrtf(var + 1e-5f) + bt.x + rv.x;
  }
  {
    const float mean = sm.y * invn, var = sq.y * invn - mean * mean;
    const float r = fmaxf(v.y, 0.f);
    o.y = gm.y * (r - mean) * rsqrtf(var + 1e-5f) + bt.y + rv.y;
  }
  {
    const float mean = sm.z * invn, var = sq.z * invn - mean * mean;
    const float r = fmaxf(v.z, 0.f);
    o.z = gm.z * (r - mean) * rsqrtf(var + 1e-5f) + bt.z + rv.z;
  }
  {
    const float mean = sm.w * invn, var = sq.w * invn - mean * mean;
    const float r = fmaxf(v.w, 0.f);
    o.w = gm.w * (r - mean) * rsqrtf(var + 1e-5f) + bt.w + rv.w;
  }
  *(float4*)&out[off] = o;
}

// ---------------------------------------------------------------------------
extern "C" void kernel_launch(void* const* d_in, const int* in_sizes, int n_in,
                              void* d_out, int out_size, void* d_ws, size_t ws_size,
                              hipStream_t stream) {
  const float*     x     = (const float*)d_in[0];
  const long long* ei    = (const long long*)d_in[1];  // [2, E] int64
  const float*     W     = (const float*)d_in[2];
  const float*     b     = (const float*)d_in[3];
  const float*     gamma = (const float*)d_in[4];
  const float*     beta  = (const float*)d_in[5];
  const float*     Wres  = (const float*)d_in[6];
  const float*     bres  = (const float*)d_in[7];

  const int n  = in_sizes[0] / IN_F;   // 100000
  const int nE = in_sizes[1] / 2;      // 600000

  // workspace: h[n*128] | res[n*128] | deg[n] | stats[256]
  float* h     = (float*)d_ws;
  float* res   = h + (size_t)n * OUT_F;
  float* deg   = res + (size_t)n * OUT_F;
  float* stats = deg + n;

  float* agg = (float*)d_out;          // agg lives in d_out; finalized in place

  // 1) dual GEMM via bf16 WMMA: h = x@W ; res = x@W_res + b_res
  dim3 ggrid((n + 63) / 64, 2);
  gcn_gemm_bf16<<<ggrid, 256, 0, stream>>>(x, W, Wres, bres, h, res, n);

  // 2) degrees (self-loop = 1) and stats buffers
  deg_init<<<(n + 255) / 256, 256, 0, stream>>>(deg, n);
  zero_stats<<<1, 256, 0, stream>>>(stats);
  deg_count<<<(nE + 255) / 256, 256, 0, stream>>>(ei + nE, deg, nE);

  // 3) agg = b + h/deg (self-loop message folded in), float4
  agg_init<<<(n * 32 + 255) / 256, 256, 0, stream>>>(h, deg, b, agg, n);

  // 4) edge messages: one wave per edge, float4 lanes, L2-resident scatter-add
  const long long scatterThreads = (long long)nE * 32;
  edge_scatter<<<(unsigned)((scatterThreads + 255) / 256), 256, 0, stream>>>(
      ei, h, deg, agg, nE);

  // 5) batch-norm statistics over relu(agg)
  bn_stats<<<(n + 255) / 256, 256, 0, stream>>>(agg, stats, n);

  // 6) fused relu + BN + residual add, float4
  finalize<<<(n * 32 + 255) / 256, 256, 0, stream>>>(
      agg, res, stats, gamma, beta, (float*)d_out, n);
}
